// TACTiS_23373212025303
// MI455X (gfx1250) — matrix operand order repaired and checked by reading, not compiled, not run on hardware
//
#include <hip/hip_runtime.h>

// ---------------------------------------------------------------------------
// TACTiS attentional-copula sampler for gfx1250 (MI455X).
// All big matmuls run on v_wmma_f32_16x16x32_bf16 (fp32 accumulate).
// GEMM: 128x64 block tile, 8 waves x (32x32) wave tiles, BK=32,
//       async global->LDS staging (ASYNCcnt) for the A tile when available.
// ---------------------------------------------------------------------------

#define DEV __device__ __forceinline__

typedef __bf16 bf16;
typedef __attribute__((ext_vector_type(16))) __bf16 v16bf;
typedef __attribute__((ext_vector_type(8)))  float  v8f;

static constexpr int Bb   = 32;    // batch
static constexpr int Vv   = 512;   // variables (sequence)
static constexpr int Dd   = 256;   // input dim
static constexpr int Ll   = 4;     // layers
static constexpr int Hh   = 8;     // heads
static constexpr int DHd  = 64;    // head dim
static constexpr int HDd  = 512;   // H*DH
static constexpr int MLPd = 1024;
static constexpr int RESd = 128;
static constexpr int Ss   = 128;   // samples
static constexpr int BV   = Bb * Vv;

struct alignas(16) Q4 { unsigned x, y, z, w; };
union FragU { Q4 q[2]; v16bf v; };
union H8U   { Q4 q; bf16 h[8]; };

static DEV v16bf ld_frag(const bf16* lo, const bf16* hi) {
  FragU u;
  u.q[0] = *(const Q4*)lo;
  u.q[1] = *(const Q4*)hi;
  return u.v;
}

static DEV v8f wmma_bf16(v16bf a, v16bf b, v8f c) {
  // D = A(16x32) * B(32x16) + C, fp32 accumulate
  return __builtin_amdgcn_wmma_f32_16x16x32_bf16(
      /*neg_a=*/false, a, /*neg_b=*/false, b,
      /*c_mod=*/(short)0, c, /*reuse_a=*/false, /*reuse_b=*/false);
}

// ---- optional CDNA5 async global->LDS path (ASYNCcnt tracked) -------------
#if defined(__has_builtin)
#if __has_builtin(__builtin_amdgcn_global_load_async_to_lds_b128) && \
    __has_builtin(__builtin_amdgcn_s_wait_asynccnt)
#define USE_ASYNC_LDS 1
#endif
#endif
#ifndef USE_ASYNC_LDS
#define USE_ASYNC_LDS 0
#endif

#if USE_ASYNC_LDS
typedef int v4i __attribute__((vector_size(16)));
static DEV void async_cp16(const bf16* g, bf16* l) {
  __builtin_amdgcn_global_load_async_to_lds_b128(
      (__attribute__((address_space(1))) v4i*)(v4i*)(void*)g,
      (__attribute__((address_space(3))) v4i*)(v4i*)(void*)l, 0, 0);
}
static DEV void async_wait0() { __builtin_amdgcn_s_wait_asynccnt(0); }
#endif

// ---------------------------------------------------------------------------
// fp32 -> bf16 cast (grid-stride)
// ---------------------------------------------------------------------------
__global__ void cast_kernel(const float* __restrict__ s, bf16* __restrict__ d, long n) {
  long i = (long)blockIdx.x * blockDim.x + threadIdx.x;
  long stride = (long)gridDim.x * blockDim.x;
  for (; i < n; i += stride) d[i] = (bf16)s[i];
}

// ---------------------------------------------------------------------------
// Generic bf16 WMMA GEMM: C[M,N] = A[M,K] * B[K,N] (+bias, opt ReLU)
//   - 128x64 tile / block, 256 threads = 8 waves, each wave a 32x32 sub-tile
//     (4 WMMAs per k-step, each A/B fragment reused twice)
//   - blockIdx.z selects a B matrix (per-head projections): B += z*strideBz,
//     bias += z*biasStrideZ, output column base += z*colStrideZ
//   - writes fp32 and/or bf16 with row stride ldc
// ---------------------------------------------------------------------------
__global__ void gemm_kernel(const bf16* __restrict__ A, const bf16* __restrict__ Bw,
                            const float* __restrict__ bias,
                            float* __restrict__ outF, bf16* __restrict__ outH,
                            int M, int N, int K, long strideBz, int biasStrideZ,
                            int colStrideZ, int ldc, int relu) {
  const int t = threadIdx.x, lane = t & 31, wave = t >> 5;
  const int wm = wave >> 1, wn = wave & 1;        // 4x2 wave grid
  const int z = blockIdx.z;
  const bf16*  Bz = Bw + (long)z * strideBz;
  const float* bz = bias ? (bias + (long)z * biasStrideZ) : nullptr;
  const int bCol    = blockIdx.x * 64;
  const int rowBase = blockIdx.y * 128;
  const int outCol  = bCol + z * colStrideZ;

  __shared__ bf16 sA[128 * 32];   // [m][k]
  __shared__ bf16 sBt[64 * 32];   // [n][k] (transposed)

  const int arow = t >> 2, acol = (t & 3) * 8;   // A tile loader (2 rows/thread)
  const int brow = t >> 3, bcol = (t & 7) * 8;   // B tile loader
  const int lm   = lane & 15;
  const int ksel = (lane < 16) ? 0 : 8;          // unused; A fragments also use kb16
  const int kb16 = (lane < 16) ? 0 : 16;         // B-fragment K half

  const bf16* aSrc0 = A + (long)(rowBase + arow) * K + acol;
  const bf16* aSrc1 = A + (long)(rowBase + arow + 64) * K + acol;

  v8f acc00 = {}, acc01 = {}, acc10 = {}, acc11 = {};
  for (int k0 = 0; k0 < K; k0 += 32) {
#if USE_ASYNC_LDS
    async_cp16(aSrc0 + k0, sA + arow * 32 + acol);
    async_cp16(aSrc1 + k0, sA + (arow + 64) * 32 + acol);
#else
    *(Q4*)(sA + arow * 32 + acol)        = *(const Q4*)(aSrc0 + k0);
    *(Q4*)(sA + (arow + 64) * 32 + acol) = *(const Q4*)(aSrc1 + k0);
#endif
    {
      const bf16* bSrc = Bz + (long)(k0 + brow) * N + bCol + bcol;
      H8U u;
      u.q = *(const Q4*)bSrc;
#pragma unroll
      for (int i = 0; i < 8; ++i) sBt[(bcol + i) * 32 + brow] = u.h[i];
      if (k0 + 32 < K) {                  // prefetch next k-tile
        __builtin_prefetch(bSrc + 32 * N, 0, 1);
        __builtin_prefetch(aSrc0 + k0 + 32, 0, 1);
      }
    }
#if USE_ASYNC_LDS
    async_wait0();
#endif
    __syncthreads();

    v16bf af0 = ld_frag(sA + (wm * 32 + lm) * 32 + kb16,
                        sA + (wm * 32 + lm) * 32 + kb16 + 8);
    v16bf af1 = ld_frag(sA + (wm * 32 + 16 + lm) * 32 + kb16,
                        sA + (wm * 32 + 16 + lm) * 32 + kb16 + 8);
    v16bf b0 = ld_frag(sBt + (wn * 32 + lm) * 32 + kb16,
                       sBt + (wn * 32 + lm) * 32 + kb16 + 8);
    v16bf b1 = ld_frag(sBt + (wn * 32 + 16 + lm) * 32 + kb16,
                       sBt + (wn * 32 + 16 + lm) * 32 + kb16 + 8);
    acc00 = wmma_bf16(af0, b0, acc00);
    acc01 = wmma_bf16(af0, b1, acc01);
    acc10 = wmma_bf16(af1, b0, acc10);
    acc11 = wmma_bf16(af1, b1, acc11);
    __syncthreads();
  }

#pragma unroll
  for (int mi = 0; mi < 2; ++mi) {
#pragma unroll
    for (int ni = 0; ni < 2; ++ni) {
      v8f acc = mi ? (ni ? acc11 : acc10) : (ni ? acc01 : acc00);
      const int nl = wn * 32 + ni * 16 + lm;   // local column in 64-wide tile
      const int n  = outCol + nl;
      const float bval = bz ? bz[bCol + nl] : 0.f;
#pragma unroll
      for (int r = 0; r < 8; ++r) {
        const int m = rowBase + wm * 32 + mi * 16 + r + ((lane < 16) ? 0 : 8);
        float v = acc[r] + bval;
        if (relu) v = fmaxf(v, 0.f);
        if (outF) outF[(long)m * ldc + n] = v;
        if (outH) outH[(long)m * ldc + n] = (bf16)v;
      }
    }
  }
}

// ---------------------------------------------------------------------------
// Flash attention, one (b,h,64-query tile) per 128-thread WG (4 waves).
// Q from att (bf16 mirror), K/V laid out [B,V,H,DH]. Online softmax.
// ---------------------------------------------------------------------------
__global__ void attn_kernel(const bf16* __restrict__ Q, const bf16* __restrict__ Ks,
                            const bf16* __restrict__ Vs, float* __restrict__ ctx) {
  const int t = threadIdx.x, lane = t & 31, wave = t >> 5;
  const int b = blockIdx.z, h = blockIdx.y;
  const int qbase = blockIdx.x * 64 + wave * 16;
  const long rowOff = (long)b * Vv * HDd + h * DHd;
  const bf16* Qb = Q  + rowOff;
  const bf16* Kb = Ks + rowOff;
  const bf16* Vb = Vs + rowOff;

  __shared__ bf16 sVt[DHd * 32];   // V tile transposed: [dh][key]
  __shared__ bf16 sP[4][16 * 32];  // per-wave P tile:   [m][k]

  const int lm   = lane & 15;
  const int ksel = (lane < 16) ? 0 : 8;
  const int kb16 = (lane < 16) ? 0 : 16;

  // Q A-fragments for dh chunks 0..31 and 32..63 (loop-invariant)
  const bf16* qrow = Qb + (long)(qbase + lm) * HDd;
  v16bf aq0 = ld_frag(qrow + kb16,      qrow + kb16 + 8);
  v16bf aq1 = ld_frag(qrow + 32 + kb16, qrow + 32 + kb16 + 8);

  float mrow[8], lrow[8], fac[8];
#pragma unroll
  for (int r = 0; r < 8; ++r) { mrow[r] = -1e30f; lrow[r] = 0.f; }
  v8f O0 = {}, O1 = {}, O2 = {}, O3 = {};

  const int vkey = t >> 2;         // 0..31
  const int vdg  = (t & 3) * 16;   // 0,16,32,48
  bf16* myP = sP[wave];

  for (int j = 0; j < Vv / 32; ++j) {
    {  // cooperative V-tile transpose into LDS
      const bf16* src = Vb + (long)(j * 32 + vkey) * HDd + vdg;
      H8U a0, a1;
      a0.q = *(const Q4*)src;
      a1.q = *(const Q4*)(src + 8);
#pragma unroll
      for (int i = 0; i < 8; ++i) {
        sVt[(vdg + i) * 32 + vkey]     = a0.h[i];
        sVt[(vdg + 8 + i) * 32 + vkey] = a1.h[i];
      }
    }
    __syncthreads();

    // scores for 32 keys: two 16x16 tiles, K-dim 64 in two chunks
    v8f s0 = {}, s1 = {};
    {
      const bf16* k0p = Kb + (long)(j * 32 + lm) * HDd;
      const bf16* k1p = Kb + (long)(j * 32 + 16 + lm) * HDd;
      v16bf kf;
      kf = ld_frag(k0p + kb16,      k0p + kb16 + 8);      s0 = wmma_bf16(aq0, kf, s0);
      kf = ld_frag(k0p + 32 + kb16, k0p + 32 + kb16 + 8); s0 = wmma_bf16(aq1, kf, s0);
      kf = ld_frag(k1p + kb16,      k1p + kb16 + 8);      s1 = wmma_bf16(aq0, kf, s1);
      kf = ld_frag(k1p + 32 + kb16, k1p + 32 + kb16 + 8); s1 = wmma_bf16(aq1, kf, s1);
    }

    // online softmax (row reductions across the 16-lane halves)
#pragma unroll
    for (int r = 0; r < 8; ++r) {
      float x0 = s0[r] * 0.125f, x1 = s1[r] * 0.125f;   // dh^-0.5
      float tm = fmaxf(x0, x1);
#pragma unroll
      for (int msk = 8; msk >= 1; msk >>= 1) tm = fmaxf(tm, __shfl_xor(tm, msk, 32));
      float nm = fmaxf(mrow[r], tm);
      float f  = __expf(mrow[r] - nm);
      float p0 = __expf(x0 - nm), p1 = __expf(x1 - nm);
      float ps = p0 + p1;
#pragma unroll
      for (int msk = 8; msk >= 1; msk >>= 1) ps += __shfl_xor(ps, msk, 32);
      lrow[r] = lrow[r] * f + ps;
      mrow[r] = nm;
      fac[r]  = f;
      const int mloc = r + ((lane < 16) ? 0 : 8);
      myP[mloc * 32 + lm]      = (bf16)p0;   // transpose C-layout -> A-layout
      myP[mloc * 32 + 16 + lm] = (bf16)p1;
    }
#pragma unroll
    for (int r = 0; r < 8; ++r) {
      O0[r] *= fac[r]; O1[r] *= fac[r]; O2[r] *= fac[r]; O3[r] *= fac[r];
    }

    v16bf pf = ld_frag(myP + lm * 32 + kb16, myP + lm * 32 + kb16 + 8);
    v16bf vf;
    vf = ld_frag(sVt + (0 * 16 + lm) * 32 + kb16, sVt + (0 * 16 + lm) * 32 + kb16 + 8);
    O0 = wmma_bf16(pf, vf, O0);
    vf = ld_frag(sVt + (1 * 16 + lm) * 32 + kb16, sVt + (1 * 16 + lm) * 32 + kb16 + 8);
    O1 = wmma_bf16(pf, vf, O1);
    vf = ld_frag(sVt + (2 * 16 + lm) * 32 + kb16, sVt + (2 * 16 + lm) * 32 + kb16 + 8);
    O2 = wmma_bf16(pf, vf, O2);
    vf = ld_frag(sVt + (3 * 16 + lm) * 32 + kb16, sVt + (3 * 16 + lm) * 32 + kb16 + 8);
    O3 = wmma_bf16(pf, vf, O3);
    __syncthreads();
  }

#pragma unroll
  for (int r = 0; r < 8; ++r) {
    const int  mg  = qbase + r + ((lane < 16) ? 0 : 8);
    const float iv = 1.f / lrow[r];
    float* dst = ctx + ((long)b * Vv + mg) * HDd + h * DHd + lm;
    dst[0]  = O0[r] * iv;
    dst[16] = O1[r] * iv;
    dst[32] = O2[r] * iv;
    dst[48] = O3[r] * iv;
  }
}

// ---------------------------------------------------------------------------
// residual add + LayerNorm over HD=512 (one row per 128-thread block)
// ---------------------------------------------------------------------------
__global__ void add_ln_kernel(const float* __restrict__ x, const float* __restrict__ dx,
                              const float* __restrict__ g, const float* __restrict__ bb,
                              float* __restrict__ outF, bf16* __restrict__ outH) {
  const int row = blockIdx.x, t = threadIdx.x, lane = t & 31, wave = t >> 5;
  const long off = (long)row * HDd;
  float v[4], s = 0.f, sq = 0.f;
#pragma unroll
  for (int i = 0; i < 4; ++i) {
    const int idx = t + i * 128;
    const float val = x[off + idx] + dx[off + idx];
    v[i] = val; s += val; sq += val * val;
  }
#pragma unroll
  for (int m = 16; m >= 1; m >>= 1) {
    s  += __shfl_xor(s, m, 32);
    sq += __shfl_xor(sq, m, 32);
  }
  __shared__ float rs_[4], rq_[4];
  if (lane == 0) { rs_[wave] = s; rq_[wave] = sq; }
  __syncthreads();
  s  = rs_[0] + rs_[1] + rs_[2] + rs_[3];
  sq = rq_[0] + rq_[1] + rq_[2] + rq_[3];
  const float mean = s * (1.f / HDd);
  const float var  = sq * (1.f / HDd) - mean * mean;
  const float rsd  = rsqrtf(var + 1e-5f);
#pragma unroll
  for (int i = 0; i < 4; ++i) {
    const int idx = t + i * 128;
    const float y = (v[i] - mean) * rsd * g[idx] + bb[idx];
    outF[off + idx] = y;
    outH[off + idx] = (bf16)y;
  }
}

// ---------------------------------------------------------------------------
// categorical sampling over RES=128 bins + dequantize; one row per block,
// one sample per thread (S == RES == 128)
// ---------------------------------------------------------------------------
static DEV unsigned pcg(unsigned x) {
  x = x * 747796405u + 2891336453u;
  unsigned w = ((x >> ((x >> 28) + 4u)) ^ x) * 277803737u;
  return (w >> 22) ^ w;
}

__global__ void sample_kernel(const float* __restrict__ logits, float* __restrict__ out) {
  const int row = blockIdx.x, t = threadIdx.x;
  __shared__ float buf[RESd];
  __shared__ float redm[RESd];
  const float lg = logits[(long)row * RESd + t];
  redm[t] = lg; __syncthreads();
  for (int s = 64; s >= 1; s >>= 1) {
    if (t < s) redm[t] = fmaxf(redm[t], redm[t + s]);
    __syncthreads();
  }
  const float mx = redm[0];
  __syncthreads();
  buf[t] = __expf(lg - mx);
  __syncthreads();
  for (int offv = 1; offv < RESd; offv <<= 1) {   // inclusive scan -> CDF
    const float add = (t >= offv) ? buf[t - offv] : 0.f;
    __syncthreads();
    buf[t] += add;
    __syncthreads();
  }
  const float total = buf[RESd - 1];
  unsigned st = (unsigned)(row * 9781 + t * 6271) ^ 0x9E3779B9u;
  const unsigned r1 = pcg(st);
  const unsigned r2 = pcg(st ^ 0x85EBCA6Bu);
  const float u1 = (r1 >> 8) * (1.f / 16777216.f) * total;
  int lo = 0, hi = RESd - 1;
  while (lo < hi) { const int mid = (lo + hi) >> 1; if (buf[mid] <= u1) lo = mid + 1; else hi = mid; }
  const float u2 = (r2 >> 8) * (1.f / 16777216.f);
  out[(long)row * Ss + t] = ((float)lo + u2) * (1.f / RESd);
}

// ---------------------------------------------------------------------------
extern "C" void kernel_launch(void* const* d_in, const int* in_sizes, int n_in,
                              void* d_out, int out_size, void* d_ws, size_t ws_size,
                              hipStream_t stream) {
  (void)in_sizes; (void)n_in; (void)out_size; (void)ws_size;
  const float* flow = (const float*)d_in[0];
  const float* Wsh  = (const float*)d_in[1];
  const float* bsh  = (const float*)d_in[2];
  const float* Wk   = (const float*)d_in[3];
  const float* bk   = (const float*)d_in[4];
  const float* Wv   = (const float*)d_in[5];
  const float* bv   = (const float*)d_in[6];
  const float* ln1g = (const float*)d_in[7];
  const float* ln1b = (const float*)d_in[8];
  const float* ln2g = (const float*)d_in[9];
  const float* ln2b = (const float*)d_in[10];
  const float* W1   = (const float*)d_in[11];
  const float* b1   = (const float*)d_in[12];
  const float* W2   = (const float*)d_in[13];
  const float* b2   = (const float*)d_in[14];
  const float* W3   = (const float*)d_in[15];
  const float* b3   = (const float*)d_in[16];
  const float* Wd   = (const float*)d_in[17];
  const float* bd   = (const float*)d_in[18];
  float* out = (float*)d_out;

  char* ws = (char*)d_ws;
  size_t off = 0;
  auto alloc = [&](size_t bytes) -> void* {
    void* p = ws + off;
    off = (off + bytes + 255) & ~(size_t)255;
    return p;
  };
  bf16*  flowH = (bf16*) alloc((size_t)BV * Dd * 2);
  float* attF  = (float*)alloc((size_t)BV * HDd * 4);
  bf16*  attH  = (bf16*) alloc((size_t)BV * HDd * 2);
  float* resF  = (float*)alloc((size_t)BV * HDd * 4);   // ctx, then ff3
  bf16*  ff1H  = (bf16*) alloc((size_t)BV * MLPd * 2);
  bf16*  ff2H  = (bf16*) alloc((size_t)BV * MLPd * 2);
  bf16*  keysH = (bf16*) alloc((size_t)BV * HDd * 2);   // [B,V,H,DH]
  bf16*  valsH = (bf16*) alloc((size_t)BV * HDd * 2);
  float* logF  = (float*)alloc((size_t)BV * RESd * 4);
  bf16*  WshH  = (bf16*) alloc((size_t)Dd * HDd * 2);
  bf16*  WkH   = (bf16*) alloc((size_t)Ll * Hh * Dd * DHd * 2);
  bf16*  WvH   = (bf16*) alloc((size_t)Ll * Hh * Dd * DHd * 2);
  bf16*  W1H   = (bf16*) alloc((size_t)Ll * HDd * MLPd * 2);
  bf16*  W2H   = (bf16*) alloc((size_t)Ll * MLPd * MLPd * 2);
  bf16*  W3H   = (bf16*) alloc((size_t)Ll * MLPd * HDd * 2);
  bf16*  WdH   = (bf16*) alloc((size_t)HDd * RESd * 2);

  auto cast = [&](const float* s, bf16* d, long n) {
    int blocks = (int)((n + 255) / 256);
    if (blocks > 2048) blocks = 2048;
    cast_kernel<<<blocks, 256, 0, stream>>>(s, d, n);
  };
  cast(flow, flowH, (long)BV * Dd);
  cast(Wsh,  WshH,  (long)Dd * HDd);
  cast(Wk,   WkH,   (long)Ll * Hh * Dd * DHd);
  cast(Wv,   WvH,   (long)Ll * Hh * Dd * DHd);
  cast(W1,   W1H,   (long)Ll * HDd * MLPd);
  cast(W2,   W2H,   (long)Ll * MLPd * MLPd);
  cast(W3,   W3H,   (long)Ll * MLPd * HDd);
  cast(Wd,   WdH,   (long)HDd * RESd);

  // att = flow @ W_shift + b_shift   (fp32 + bf16 mirror)
  gemm_kernel<<<dim3(HDd / 64, BV / 128, 1), 256, 0, stream>>>(
      flowH, WshH, bsh, attF, attH, BV, HDd, Dd, 0, 0, 0, HDd, 0);

  for (int l = 0; l < Ll; ++l) {
    // per-head K/V projections: gridDim.z = H, output [B,V,H,DH]
    gemm_kernel<<<dim3(1, BV / 128, Hh), 256, 0, stream>>>(
        flowH, WkH + (long)l * Hh * Dd * DHd, bk + (long)l * Hh * DHd,
        nullptr, keysH, BV, DHd, Dd, (long)Dd * DHd, DHd, DHd, HDd, 0);
    gemm_kernel<<<dim3(1, BV / 128, Hh), 256, 0, stream>>>(
        flowH, WvH + (long)l * Hh * Dd * DHd, bv + (long)l * Hh * DHd,
        nullptr, valsH, BV, DHd, Dd, (long)Dd * DHd, DHd, DHd, HDd, 0);

    attn_kernel<<<dim3(Vv / 64, Hh, Bb), 128, 0, stream>>>(attH, keysH, valsH, resF);
    add_ln_kernel<<<BV, 128, 0, stream>>>(attF, resF, ln1g + (long)l * HDd,
                                          ln1b + (long)l * HDd, attF, attH);

    gemm_kernel<<<dim3(MLPd / 64, BV / 128, 1), 256, 0, stream>>>(
        attH, W1H + (long)l * HDd * MLPd, b1 + (long)l * MLPd,
        nullptr, ff1H, BV, MLPd, HDd, 0, 0, 0, MLPd, 1);
    gemm_kernel<<<dim3(MLPd / 64, BV / 128, 1), 256, 0, stream>>>(
        ff1H, W2H + (long)l * MLPd * MLPd, b2 + (long)l * MLPd,
        nullptr, ff2H, BV, MLPd, MLPd, 0, 0, 0, MLPd, 1);
    gemm_kernel<<<dim3(HDd / 64, BV / 128, 1), 256, 0, stream>>>(
        ff2H, W3H + (long)l * MLPd * HDd, b3 + (long)l * HDd,
        resF, nullptr, BV, HDd, MLPd, 0, 0, 0, HDd, 0);
    add_ln_kernel<<<BV, 128, 0, stream>>>(attF, resF, ln2g + (long)l * HDd,
                                          ln2b + (long)l * HDd, attF, attH);
  }

  // logits = att @ W_dist + b_dist
  gemm_kernel<<<dim3(RESd / 64, BV / 128, 1), 256, 0, stream>>>(
      attH, WdH, bd, logF, nullptr, BV, RESd, HDd, 0, 0, 0, RESd, 0);

  sample_kernel<<<BV, RESd, 0, stream>>>(logF, out);
}
